// MAE_64854006169690
// MI455X (gfx1250) — compile-verified, matrix operand order checked by reading, not run on hardware
//
#include <hip/hip_runtime.h>
#include <hip/hip_bf16.h>
#include <math.h>

// ---------------- CDNA5 WMMA types & helpers ----------------
typedef __attribute__((ext_vector_type(16))) __bf16          v16bf;
typedef __attribute__((ext_vector_type(8)))  float           v8f;
typedef __attribute__((ext_vector_type(8)))  unsigned short  ushx8;
typedef __attribute__((ext_vector_type(4)))  unsigned int    uix4;
typedef unsigned short u16;

union FragU { v16bf v; ushx8 h[2]; uix4 w[2]; };

// f32 -> bf16, round-half-up (1 VALU op per element vs 2 for full RNE)
__device__ __forceinline__ unsigned int rhu(unsigned int u) {
    return u + 0x8000u;
}
__device__ __forceinline__ unsigned short f2bf(float f) {
    return (unsigned short)(rhu(__builtin_bit_cast(unsigned int, f)) >> 16);
}
// pack two f32 -> two bf16 in one dword: {hi16(b), hi16(a)} via v_perm_b32
__device__ __forceinline__ unsigned int pack2(float a, float b) {
    unsigned int ua = rhu(__builtin_bit_cast(unsigned int, a));
    unsigned int ub = rhu(__builtin_bit_cast(unsigned int, b));
    return __builtin_amdgcn_perm(ub, ua, 0x07060302u);
}

// Per-lane K-base for 16-bit WMMA A/B fragments (wave32):
// lanes 0-15 hold K 0..7 & 16..23, lanes 16-31 hold K 8..15 & 24..31.
// Fragment = two contiguous 8-element (16B) chunks at kb0 and kb0+16.
__device__ __forceinline__ int kbase(int lane) { return (lane & 16) ? 8 : 0; }

// ---------------- WMMA GEMM: C = act(A @ W + bias) ----------------
// A: [M,K] f32, W: [K,N] f32, bias: [N], C: [M,N] f32.
// Block: 256 thr (8 waves). Block tile 128x128, wave tile 32x64 (2x4 wmma),
// K-step 32, bf16 LDS staging (A row-major, W transposed).
// Software pipeline: global loads for tile k+32 issued before the wmma phase
// of tile k, consumed (convert+LDS store) one iteration later.
__global__ __launch_bounds__(256)
void gemm_bias_kernel(const float* __restrict__ A, const float* __restrict__ W,
                      const float* __restrict__ bias, float* __restrict__ C,
                      int M, int N, int K, int relu)
{
    __shared__ u16 sA [128][40];   // [m][k]   row stride 80B (16B mult)
    __shared__ u16 sBt[128][40];   // [n][k]   transposed weight tile
    const int tid   = threadIdx.x;
    const int lane  = tid & 31;
    const int wave  = tid >> 5;
    const int waveM = wave >> 1;   // 0..3
    const int waveN = wave & 1;    // 0..1
    const int bM    = blockIdx.y * 128;
    const int bN    = blockIdx.x * 128;
    const int kb0   = kbase(lane);

    float2 rA[8];                  // register-staged A tile slice
    float  rW[16];                 // register-staged W tile slice

    auto load_regs = [&](int k0) {
        #pragma unroll
        for (int j = 0; j < 8; j++) {          // A: coalesced float2 along k
            int i = tid + j * 256;
            int r = i >> 4, cp = (i & 15) << 1;
            rA[j] = *(const float2*)&A[(size_t)(bM + r) * K + k0 + cp];
        }
        #pragma unroll
        for (int j = 0; j < 8; j++) {          // W: coalesced along n, pair of k-rows
            int i = tid + j * 256;
            int c = i & 127, rp = (i >> 7) << 1;
            rW[2 * j]     = W[(size_t)(k0 + rp)     * N + bN + c];
            rW[2 * j + 1] = W[(size_t)(k0 + rp + 1) * N + bN + c];
        }
    };
    auto store_lds = [&]() {
        #pragma unroll
        for (int j = 0; j < 8; j++) {
            int i = tid + j * 256;
            int r = i >> 4, cp = (i & 15) << 1;
            *(unsigned int*)&sA[r][cp] = pack2(rA[j].x, rA[j].y);
        }
        #pragma unroll
        for (int j = 0; j < 8; j++) {
            int i = tid + j * 256;
            int c = i & 127, rp = (i >> 7) << 1;
            *(unsigned int*)&sBt[c][rp] = pack2(rW[2 * j], rW[2 * j + 1]);
        }
    };

    v8f acc[2][4];
    #pragma unroll
    for (int s = 0; s < 2; s++)
        #pragma unroll
        for (int t = 0; t < 4; t++) acc[s][t] = {};

    load_regs(0);
    for (int k0 = 0; k0 < K; k0 += 32) {
        store_lds();
        __syncthreads();
        if (k0 + 32 < K) load_regs(k0 + 32);   // overlap next-tile loads with wmma

        FragU af[2], bf[4];
        #pragma unroll
        for (int s = 0; s < 2; s++) {
            int m = waveM * 32 + s * 16 + (lane & 15);
            af[s].h[0] = *(const ushx8*)&sA[m][kb0];
            af[s].h[1] = *(const ushx8*)&sA[m][kb0 + 16];
        }
        #pragma unroll
        for (int t = 0; t < 4; t++) {
            int n = waveN * 64 + t * 16 + (lane & 15);
            bf[t].h[0] = *(const ushx8*)&sBt[n][kb0];
            bf[t].h[1] = *(const ushx8*)&sBt[n][kb0 + 16];
        }
        #pragma unroll
        for (int s = 0; s < 2; s++)
            #pragma unroll
            for (int t = 0; t < 4; t++)
                acc[s][t] = __builtin_amdgcn_wmma_f32_16x16x32_bf16(
                    false, af[s].v, false, bf[t].v, (short)0, acc[s][t], false, false);
        __syncthreads();
    }

    const int rh = (lane >> 4) << 3;  // C layout: row = g + (lane<16 ? 0 : 8)
    #pragma unroll
    for (int s = 0; s < 2; s++)
        #pragma unroll
        for (int t = 0; t < 4; t++)
            #pragma unroll
            for (int g = 0; g < 8; g++) {
                int row = bM + waveM * 32 + s * 16 + g + rh;
                int col = bN + waveN * 64 + t * 16 + (lane & 15);
                float v = acc[s][t][g] + bias[col];
                if (relu) v = fmaxf(v, 0.0f);
                C[(size_t)row * N + col] = v;
            }
}

// ---------------- Flash attention (WMMA, online softmax) ----------------
// q,k,v,o: [B*H, NSEQ, DK] f32 (contiguous reinterpretation of [B,n,d]).
// Block: 128 thr (4 waves); each wave owns 16 query rows; block shares K/V.
template <int DK, int NSEQ>
__global__ __launch_bounds__(128)
void attn_kernel(const float* __restrict__ Qg, const float* __restrict__ Kg,
                 const float* __restrict__ Vg, float* __restrict__ Og, float scale)
{
    constexpr int KP = DK + 8;          // sK row stride (16B multiple)
    __shared__ u16 sK [32][KP];         // [keyrow][feat]   (B-frag for S)
    __shared__ u16 sVt[DK][40];         // [feat][keyrow]   (B-frag for O)
    __shared__ u16 sP [4][16][40];      // per-wave P tile  (A-frag for O)

    const int tid  = threadIdx.x;
    const int lane = tid & 31;
    const int wave = tid >> 5;
    const size_t bh = blockIdx.y;
    const int q0 = blockIdx.x * 64 + wave * 16;
    const int kb0 = kbase(lane);

    const float* Q  = Qg + bh * (size_t)NSEQ * DK;
    const float* Kp = Kg + bh * (size_t)NSEQ * DK;
    const float* Vp = Vg + bh * (size_t)NSEQ * DK;
    float*       O  = Og + bh * (size_t)NSEQ * DK;

    constexpr int NQF = DK / 32;   // A fragments over feature dim
    constexpr int NOF = DK / 16;   // output col tiles
    constexpr int NIT = DK / 8;    // staging iterations per tile

    FragU qf[NQF];                 // Q pre-scaled by 1/sqrt(dk)
    {
        int m = q0 + (lane & 15);
        #pragma unroll
        for (int f = 0; f < NQF; f++) {
            float a[8], b[8];
            *(float4*)&a[0] = *(const float4*)&Q[(size_t)m * DK + f * 32 + kb0];
            *(float4*)&a[4] = *(const float4*)&Q[(size_t)m * DK + f * 32 + kb0 + 4];
            *(float4*)&b[0] = *(const float4*)&Q[(size_t)m * DK + f * 32 + kb0 + 16];
            *(float4*)&b[4] = *(const float4*)&Q[(size_t)m * DK + f * 32 + kb0 + 20];
            #pragma unroll
            for (int j = 0; j < 4; j++) {
                qf[f].w[0][j] = pack2(a[2*j] * scale, a[2*j+1] * scale);
                qf[f].w[1][j] = pack2(b[2*j] * scale, b[2*j+1] * scale);
            }
        }
    }

    v8f oacc[NOF];
    #pragma unroll
    for (int t = 0; t < NOF; t++) oacc[t] = {};
    float rowM[8], rowL[8];
    #pragma unroll
    for (int g = 0; g < 8; g++) { rowM[g] = -1e30f; rowL[g] = 0.0f; }

    for (int kb = 0; kb < NSEQ; kb += 32) {
        // two-phase staging: clustered loads into regs, then convert+store
        float2 rK[NIT];
        float  rV[2 * NIT];
        #pragma unroll
        for (int j = 0; j < NIT; j++) {
            int i = tid + j * 128;
            int r = i / (DK / 2), cp = (i % (DK / 2)) * 2;
            rK[j] = *(const float2*)&Kp[(size_t)(kb + r) * DK + cp];
        }
        #pragma unroll
        for (int j = 0; j < NIT; j++) {
            int i = tid + j * 128;
            int c = i & (DK - 1), rp = (i / DK) << 1;
            rV[2 * j]     = Vp[(size_t)(kb + rp)     * DK + c];
            rV[2 * j + 1] = Vp[(size_t)(kb + rp + 1) * DK + c];
        }
        #pragma unroll
        for (int j = 0; j < NIT; j++) {
            int i = tid + j * 128;
            int r = i / (DK / 2), cp = (i % (DK / 2)) * 2;
            *(unsigned int*)&sK[r][cp] = pack2(rK[j].x, rK[j].y);
        }
        #pragma unroll
        for (int j = 0; j < NIT; j++) {
            int i = tid + j * 128;
            int c = i & (DK - 1), rp = (i / DK) << 1;
            *(unsigned int*)&sVt[c][rp] = pack2(rV[2 * j], rV[2 * j + 1]);
        }
        __syncthreads();

        v8f s0 = {}, s1 = {};                           // S = Qs @ K^T (16x32)
        #pragma unroll
        for (int f = 0; f < NQF; f++) {
            FragU b0, b1;
            int n = lane & 15;
            b0.h[0] = *(const ushx8*)&sK[n]     [f * 32 + kb0];
            b0.h[1] = *(const ushx8*)&sK[n]     [f * 32 + kb0 + 16];
            b1.h[0] = *(const ushx8*)&sK[n + 16][f * 32 + kb0];
            b1.h[1] = *(const ushx8*)&sK[n + 16][f * 32 + kb0 + 16];
            s0 = __builtin_amdgcn_wmma_f32_16x16x32_bf16(false, qf[f].v, false, b0.v, (short)0, s0, false, false);
            s1 = __builtin_amdgcn_wmma_f32_16x16x32_bf16(false, qf[f].v, false, b1.v, (short)0, s1, false, false);
        }

        float corr[8], p0[8], p1[8];
        #pragma unroll
        for (int g = 0; g < 8; g++) {                   // online softmax (row lives in half-wave)
            float m = fmaxf(s0[g], s1[g]);
            #pragma unroll
            for (int off = 1; off < 16; off <<= 1) m = fmaxf(m, __shfl_xor(m, off, 32));
            float nm = fmaxf(rowM[g], m);
            corr[g] = __expf(rowM[g] - nm);
            rowM[g] = nm;
            p0[g] = __expf(s0[g] - nm);
            p1[g] = __expf(s1[g] - nm);
            float rs = p0[g] + p1[g];
            #pragma unroll
            for (int off = 1; off < 16; off <<= 1) rs += __shfl_xor(rs, off, 32);
            rowL[g] = rowL[g] * corr[g] + rs;
        }
        #pragma unroll
        for (int t = 0; t < NOF; t++)
            #pragma unroll
            for (int g = 0; g < 8; g++) oacc[t][g] *= corr[g];

        {   // P: C-layout -> per-wave LDS tile (row-major)
            int n = lane & 15, rh = (lane >> 4) << 3;
            #pragma unroll
            for (int g = 0; g < 8; g++) {
                sP[wave][g + rh][n]      = f2bf(p0[g]);
                sP[wave][g + rh][n + 16] = f2bf(p1[g]);
            }
        }
        FragU pf;
        {
            int m = lane & 15;
            pf.h[0] = *(const ushx8*)&sP[wave][m][kb0];
            pf.h[1] = *(const ushx8*)&sP[wave][m][kb0 + 16];
        }
        #pragma unroll
        for (int t = 0; t < NOF; t++) {                 // O += P @ V
            FragU bv;
            int n = lane & 15;
            bv.h[0] = *(const ushx8*)&sVt[t * 16 + n][kb0];
            bv.h[1] = *(const ushx8*)&sVt[t * 16 + n][kb0 + 16];
            oacc[t] = __builtin_amdgcn_wmma_f32_16x16x32_bf16(false, pf.v, false, bv.v, (short)0, oacc[t], false, false);
        }
        __syncthreads();
    }

    int rh = (lane >> 4) << 3;
    #pragma unroll
    for (int g = 0; g < 8; g++) {
        float inv = 1.0f / rowL[g];
        int row = q0 + g + rh;
        #pragma unroll
        for (int t = 0; t < NOF; t++)
            O[(size_t)row * DK + t * 16 + (lane & 15)] = oacc[t][g] * inv;
    }
}

// ---------------- fused residual-add + LayerNorm ----------------
__global__ __launch_bounds__(256)
void add_ln_kernel(const float* __restrict__ X, const float* __restrict__ R,
                   const float* __restrict__ sc, const float* __restrict__ bs,
                   float* __restrict__ out, int d)
{
    __shared__ float red[256];
    const int row = blockIdx.x;
    const float* x = X + (size_t)row * d;
    const float* r = R + (size_t)row * d;
    float* o = out + (size_t)row * d;

    float s = 0.0f;
    for (int c = threadIdx.x; c < d; c += 256) s += x[c] + r[c];
    red[threadIdx.x] = s; __syncthreads();
    for (int st = 128; st > 0; st >>= 1) {
        if (threadIdx.x < st) red[threadIdx.x] += red[threadIdx.x + st];
        __syncthreads();
    }
    float mu = red[0] / d; __syncthreads();

    float v = 0.0f;
    for (int c = threadIdx.x; c < d; c += 256) { float t = x[c] + r[c] - mu; v += t * t; }
    red[threadIdx.x] = v; __syncthreads();
    for (int st = 128; st > 0; st >>= 1) {
        if (threadIdx.x < st) red[threadIdx.x] += red[threadIdx.x + st];
        __syncthreads();
    }
    float rstd = rsqrtf(red[0] / d + 1e-5f);
    for (int c = threadIdx.x; c < d; c += 256)
        o[c] = (x[c] + r[c] - mu) * rstd * sc[c] + bs[c];
}

// ---------------- small elementwise kernels ----------------
__global__ void posenc_kernel(float* __restrict__ out, int maxlen, int d)
{
    long long t = blockIdx.x * 256LL + threadIdx.x;
    if (t >= (long long)maxlen * d) return;
    int pos = (int)(t / d), i = (int)(t % d);
    float angle = __powf(10000.0f, (float)(2 * (i / 2)) / (float)d);
    float e = (float)pos / angle;
    out[t] = (i & 1) ? __cosf(e) : __sinf(e);
}

__global__ void gather_rows_kernel(const float* __restrict__ x, const int* __restrict__ idx,
                                   float* __restrict__ out, int B, int nin, int nout, int d)
{
    long long t = blockIdx.x * 256LL + threadIdx.x;
    long long total = (long long)B * nout * d;
    if (t >= total) return;
    int c = (int)(t % d);
    long long r = t / d;
    int i = (int)(r % nout);
    int b = (int)(r / nout);
    out[t] = x[((long long)b * nin + idx[i]) * d + c];
}

__global__ void add_pe_kernel(float* __restrict__ h, const float* __restrict__ pe,
                              const int* __restrict__ idx, int B, int n, int d)
{
    long long t = blockIdx.x * 256LL + threadIdx.x;
    long long total = (long long)B * n * d;
    if (t >= total) return;
    int c = (int)(t % d);
    int i = (int)((t / d) % n);
    h[t] += pe[(long long)idx[i] * d + c];
}

__global__ void build_dec_kernel(const float* __restrict__ hv, const float* __restrict__ pe,
                                 const int* __restrict__ vis, const int* __restrict__ mask,
                                 const float* __restrict__ mtok, float* __restrict__ out, int B)
{
    long long t = blockIdx.x * 256LL + threadIdx.x;
    long long total = (long long)B * 1024 * 512;
    if (t >= total) return;
    int c = (int)(t % 512);
    int r = (int)((t / 512) % 1024);
    int b = (int)(t / (512LL * 1024));
    if (r < 256)
        out[t] = hv[((long long)b * 256 + r) * 512 + c] + pe[(long long)vis[r] * 512 + c];
    else
        out[t] = mtok[c] + pe[(long long)mask[r - 256] * 512 + c];
}

__global__ void copy_out_kernel(const float* __restrict__ src, float* __restrict__ dst, int B)
{
    long long t = blockIdx.x * 256LL + threadIdx.x;
    long long total = (long long)B * 768 * 1024;
    if (t >= total) return;
    int c = (int)(t % 1024);
    int j = (int)((t / 1024) % 768);
    int b = (int)(t / (768LL * 1024));
    dst[t] = src[((long long)b * 1024 + 256 + j) * 1024 + c];
}

// ---------------- host orchestration ----------------
static inline unsigned eb(long long n) { return (unsigned)((n + 255) / 256); }

extern "C" void kernel_launch(void* const* d_in, const int* in_sizes, int n_in,
                              void* d_out, int out_size, void* d_ws, size_t ws_size,
                              hipStream_t stream)
{
    (void)in_sizes; (void)n_in; (void)out_size; (void)ws_size;

    // setup_inputs() insertion-order flattening:
    const float* x        = (const float*)d_in[0];
    const int*   vis      = (const int*)  d_in[1];
    const int*   mask     = (const int*)  d_in[2];
    const float* first_w  = (const float*)d_in[3];
    const float* first_b  = (const float*)d_in[4];
    const int EI = 5;   // enc: wq bq wk bk wv bv wo bo n1s n1b n2s n2b w1 b1 w2 b2
    const float* e2d_w    = (const float*)d_in[21];
    const float* e2d_b    = (const float*)d_in[22];
    const float* mtok     = (const float*)d_in[23];
    const int DI = 24;  // dec: same 16-tensor order
    const float* fin_w    = (const float*)d_in[40];
    const float* fin_b    = (const float*)d_in[41];

    // workspace layout (floats)
    float* ws   = (float*)d_ws;
    float* pe_e = ws;                    // 1024*1024
    float* pe_d = pe_e + 1048576;        // 1024*512
    float* encA = pe_d + 524288;         // 2048*1024
    float* encB = encA + 2097152;        // 2048*1024
    float* xg   = encB + 2097152;        // 2048*1024
    float* qb   = xg   + 2097152;        // max(2048*1024, 8192*512)
    float* kb   = qb   + 4194304;
    float* vb   = kb   + 4194304;
    float* ab   = vb   + 4194304;
    float* mid  = ab   + 4194304;        // max(2048*2048, 8192*1024)
    float* tmp  = mid  + 8388608;        // max gemm-out incl. final 8192*1024
    float* decA = tmp  + 8388608;        // 8192*512
    float* decB = decA + 4194304;        // 8192*512

    auto gemm = [&](const float* A, const float* W_, const float* b_, float* C,
                    int M, int N, int K, int relu) {
        dim3 g(N / 128, M / 128);
        gemm_bias_kernel<<<g, 256, 0, stream>>>(A, W_, b_, C, M, N, K, relu);
    };

    // positional encodings
    posenc_kernel<<<eb(1024LL * 1024), 256, 0, stream>>>(pe_e, 1024, 1024);
    posenc_kernel<<<eb(1024LL * 512),  256, 0, stream>>>(pe_d, 1024, 512);

    // input projection on gathered visible tokens (gather commutes with x@W)
    gather_rows_kernel<<<eb(8LL * 256 * 1024), 256, 0, stream>>>(x, vis, xg, 8, 1024, 256, 1024);
    gemm(xg, first_w, first_b, encA, 2048, 1024, 1024, 0);
    add_pe_kernel<<<eb(8LL * 256 * 1024), 256, 0, stream>>>(encA, pe_e, vis, 8, 256, 1024);

    // ---- encoder: 12 layers, d=1024, dk=128, n=256 ----
    for (int l = 0; l < 12; l++) {
        const float* wq = (const float*)d_in[EI + 0]  + (size_t)l * 1024 * 1024;
        const float* bq = (const float*)d_in[EI + 1]  + (size_t)l * 1024;
        const float* wk = (const float*)d_in[EI + 2]  + (size_t)l * 1024 * 1024;
        const float* bk = (const float*)d_in[EI + 3]  + (size_t)l * 1024;
        const float* wv = (const float*)d_in[EI + 4]  + (size_t)l * 1024 * 1024;
        const float* bv = (const float*)d_in[EI + 5]  + (size_t)l * 1024;
        const float* wo = (const float*)d_in[EI + 6]  + (size_t)l * 1024 * 1024;
        const float* bo = (const float*)d_in[EI + 7]  + (size_t)l * 1024;
        const float* n1s= (const float*)d_in[EI + 8]  + (size_t)l * 1024;
        const float* n1b= (const float*)d_in[EI + 9]  + (size_t)l * 1024;
        const float* n2s= (const float*)d_in[EI + 10] + (size_t)l * 1024;
        const float* n2b= (const float*)d_in[EI + 11] + (size_t)l * 1024;
        const float* w1 = (const float*)d_in[EI + 12] + (size_t)l * 1024 * 2048;
        const float* b1 = (const float*)d_in[EI + 13] + (size_t)l * 2048;
        const float* w2 = (const float*)d_in[EI + 14] + (size_t)l * 2048 * 1024;
        const float* b2 = (const float*)d_in[EI + 15] + (size_t)l * 1024;

        gemm(encA, wq, bq, qb, 2048, 1024, 1024, 0);
        gemm(encA, wk, bk, kb, 2048, 1024, 1024, 0);
        gemm(encA, wv, bv, vb, 2048, 1024, 1024, 0);
        attn_kernel<128, 256><<<dim3(256 / 64, 64), 128, 0, stream>>>(
            qb, kb, vb, ab, 1.0f / sqrtf(128.0f));
        gemm(ab, wo, bo, tmp, 2048, 1024, 1024, 0);
        add_ln_kernel<<<2048, 256, 0, stream>>>(tmp, encA, n1s, n1b, encB, 1024);
        gemm(encB, w1, b1, mid, 2048, 2048, 1024, 1);
        gemm(mid,  w2, b2, tmp, 2048, 1024, 2048, 0);
        add_ln_kernel<<<2048, 256, 0, stream>>>(tmp, encB, n2s, n2b, encA, 1024);
    }

    // encoder->decoder projection + decoder input assembly
    gemm(encA, e2d_w, e2d_b, tmp, 2048, 512, 1024, 0);
    build_dec_kernel<<<eb(8LL * 1024 * 512), 256, 0, stream>>>(tmp, pe_d, vis, mask, mtok, decA, 8);

    // ---- decoder: 4 layers, d=512, dk=64, n=1024 ----
    for (int l = 0; l < 4; l++) {
        const float* wq = (const float*)d_in[DI + 0]  + (size_t)l * 512 * 512;
        const float* bq = (const float*)d_in[DI + 1]  + (size_t)l * 512;
        const float* wk = (const float*)d_in[DI + 2]  + (size_t)l * 512 * 512;
        const float* bk = (const float*)d_in[DI + 3]  + (size_t)l * 512;
        const float* wv = (const float*)d_in[DI + 4]  + (size_t)l * 512 * 512;
        const float* bv = (const float*)d_in[DI + 5]  + (size_t)l * 512;
        const float* wo = (const float*)d_in[DI + 6]  + (size_t)l * 512 * 512;
        const float* bo = (const float*)d_in[DI + 7]  + (size_t)l * 512;
        const float* n1s= (const float*)d_in[DI + 8]  + (size_t)l * 512;
        const float* n1b= (const float*)d_in[DI + 9]  + (size_t)l * 512;
        const float* n2s= (const float*)d_in[DI + 10] + (size_t)l * 512;
        const float* n2b= (const float*)d_in[DI + 11] + (size_t)l * 512;
        const float* w1 = (const float*)d_in[DI + 12] + (size_t)l * 512 * 1024;
        const float* b1 = (const float*)d_in[DI + 13] + (size_t)l * 1024;
        const float* w2 = (const float*)d_in[DI + 14] + (size_t)l * 1024 * 512;
        const float* b2 = (const float*)d_in[DI + 15] + (size_t)l * 512;

        gemm(decA, wq, bq, qb, 8192, 512, 512, 0);
        gemm(decA, wk, bk, kb, 8192, 512, 512, 0);
        gemm(decA, wv, bv, vb, 8192, 512, 512, 0);
        attn_kernel<64, 1024><<<dim3(1024 / 64, 64), 128, 0, stream>>>(
            qb, kb, vb, ab, 1.0f / sqrtf(64.0f));
        gemm(ab, wo, bo, tmp, 8192, 512, 512, 0);
        add_ln_kernel<<<8192, 256, 0, stream>>>(tmp, decA, n1s, n1b, decB, 512);
        gemm(decB, w1, b1, mid, 8192, 1024, 512, 1);
        gemm(mid,  w2, b2, tmp, 8192, 512, 1024, 0);
        add_ln_kernel<<<8192, 256, 0, stream>>>(tmp, decB, n2s, n2b, decA, 512);
    }

    // final projection + slice of masked tokens
    gemm(decA, fin_w, fin_b, tmp, 8192, 1024, 512, 0);
    copy_out_kernel<<<eb(8LL * 768 * 1024), 256, 0, stream>>>(tmp, (float*)d_out, 8);
}
